// EncoderDecoder_90838558310801
// MI455X (gfx1250) — matrix operand order repaired and checked by reading, not compiled, not run on hardware
//
#include <hip/hip_runtime.h>
#include <hip/hip_bf16.h>
#include <utility>

typedef __attribute__((ext_vector_type(16))) __bf16 v16bf;
typedef __attribute__((ext_vector_type(8)))  float  v8f;

#define VOCAB 32000
#define EDIM  512
#define HDIM  1024
#define NB    4
#define NK    8
#define NROW  32          // NB*NK
#define TMAX  64
#define CTG   4           // column tiles per wave in the WMMA GEMM
#define NEG_INF (-1e30f)

// ---------------------------------------------------------------------------
// Weight packing: W [Nout x Kin] f32 row-major  ->  bf16 packed B-tiles.
// Packed layout per (colTile ct, kChunk kc): 32 lanes x 16 bf16 contiguous.
// Lane l: column = ct*16 + (l&15); K rows = kc*32 + (l>>4)*16 + j, j=0..15.
// Contiguous k order, the same order pack_a_kernel uses for A.
// ---------------------------------------------------------------------------
__global__ void pack_b_kernel(const float* __restrict__ W, __bf16* __restrict__ dst,
                              int Nout, int Kin, int nkc) {
    int t = blockIdx.x * blockDim.x + threadIdx.x;
    int total = (Nout / 16) * nkc * 32;
    if (t >= total) return;
    int lane = t & 31;
    int kc   = (t >> 5) % nkc;
    int ct   = t / (nkc * 32);
    int col  = ct * 16 + (lane & 15);
    int kb   = kc * 32 + ((lane >> 4) << 4);
    const float* s = W + (size_t)col * Kin + kb;
    __bf16* d = dst + (size_t)t * 16;
#pragma unroll
    for (int j = 0; j < 16; ++j) d[j] = (__bf16)s[j];
}

// ---------------------------------------------------------------------------
// Activation packing: src [32 x K] f32 (optional row gather) -> bf16 A-tiles.
// Lane l: row = mt*16 + (l&15); K = kc*32 + off + {0..15},
// off = 0 for lanes 0-15, 16 for lanes 16-31  (matches packed B k order).
// ---------------------------------------------------------------------------
__global__ void pack_a_kernel(const float* __restrict__ src, const int* __restrict__ gather,
                              __bf16* __restrict__ dst, int nkc, int srcLd) {
    int t = blockIdx.x * blockDim.x + threadIdx.x;
    int total = 2 * nkc * 32;
    if (t >= total) return;
    int lane = t & 31;
    int kc   = (t >> 5) % nkc;
    int mt   = t / (nkc * 32);
    int row  = mt * 16 + (lane & 15);
    const float* s = gather ? src + (size_t)gather[row] * srcLd
                            : src + (size_t)row * srcLd;
    int kb = kc * 32 + ((lane >> 4) << 4);
    __bf16* d = dst + (size_t)t * 16;
#pragma unroll
    for (int j = 0; j < 8; ++j) d[j]     = (__bf16)s[kb + j];
#pragma unroll
    for (int j = 0; j < 8; ++j) d[8 + j] = (__bf16)s[kb + 8 + j];
}

// ---------------------------------------------------------------------------
// WMMA GEMM, register-blocked: each wave owns 1 M-tile x CTG N-tiles.
// Ping-pong fragment sets (refilled by loads, never copied) so the steady
// state is loads + 4x v_wmma with staggered s_wait_loadcnt and no v_movs.
// nkc must be even (16 or 32 here).
// ---------------------------------------------------------------------------
__global__ void __launch_bounds__(128)
gemm_wmma_kernel(const __bf16* __restrict__ pA, const __bf16* __restrict__ pB,
                 const float* __restrict__ bias, float* __restrict__ C,
                 int nColTiles, int nkc, int ldc) {
    int wave = (blockIdx.x * blockDim.x + threadIdx.x) >> 5;
    int lane = threadIdx.x & 31;
    int nGroups = nColTiles / CTG;
    if (wave >= 2 * nGroups) return;         // whole-wave uniform exit
    int mt = wave / nGroups;
    int cg = wave % nGroups;

    const __bf16* aB = pA + ((size_t)mt * nkc * 32 + lane) * 16;
    const __bf16* bB = pB + ((size_t)(cg * CTG) * nkc * 32 + lane) * 16;
    const size_t bStride = (size_t)nkc * 512;   // between adjacent col tiles

    v8f acc[CTG];
#pragma unroll
    for (int j = 0; j < CTG; ++j) acc[j] = (v8f){};

    // fragment set 0 <- chunk 0
    v16bf a0 = *(const v16bf*)aB;
    v16bf b0[CTG];
#pragma unroll
    for (int j = 0; j < CTG; ++j) b0[j] = *(const v16bf*)(bB + j * bStride);

    v16bf a1;
    v16bf b1[CTG];

    int kc = 0;
    for (; kc + 3 < nkc; kc += 2) {
        // set 1 <- chunk kc+1 ; prefetch two chunks ahead on the B streams
        a1 = *(const v16bf*)(aB + (size_t)(kc + 1) * 512);
#pragma unroll
        for (int j = 0; j < CTG; ++j) {
            b1[j] = *(const v16bf*)(bB + j * bStride + (size_t)(kc + 1) * 512);
            __builtin_prefetch(bB + j * bStride + (size_t)(kc + 4) * 512, 0, 1);
        }
#pragma unroll
        for (int j = 0; j < CTG; ++j)
            acc[j] = __builtin_amdgcn_wmma_f32_16x16x32_bf16(
                         false, a0, false, b0[j], (short)0, acc[j], false, false);
        // set 0 <- chunk kc+2 (refill by load; no register rotation)
        a0 = *(const v16bf*)(aB + (size_t)(kc + 2) * 512);
#pragma unroll
        for (int j = 0; j < CTG; ++j)
            b0[j] = *(const v16bf*)(bB + j * bStride + (size_t)(kc + 2) * 512);
#pragma unroll
        for (int j = 0; j < CTG; ++j)
            acc[j] = __builtin_amdgcn_wmma_f32_16x16x32_bf16(
                         false, a1, false, b1[j], (short)0, acc[j], false, false);
    }
    // tail: chunks nkc-2 (in set 0) and nkc-1
    a1 = *(const v16bf*)(aB + (size_t)(nkc - 1) * 512);
#pragma unroll
    for (int j = 0; j < CTG; ++j)
        b1[j] = *(const v16bf*)(bB + j * bStride + (size_t)(nkc - 1) * 512);
#pragma unroll
    for (int j = 0; j < CTG; ++j)
        acc[j] = __builtin_amdgcn_wmma_f32_16x16x32_bf16(
                     false, a0, false, b0[j], (short)0, acc[j], false, false);
#pragma unroll
    for (int j = 0; j < CTG; ++j)
        acc[j] = __builtin_amdgcn_wmma_f32_16x16x32_bf16(
                     false, a1, false, b1[j], (short)0, acc[j], false, false);

    int rb = mt * 16 + ((lane >> 4) << 3);   // C: VGPR v -> M = v + 8*(lane>=16)
#pragma unroll
    for (int j = 0; j < CTG; ++j) {
        int col  = (cg * CTG + j) * 16 + (lane & 15);
        float bv = bias ? bias[col] : 0.0f;
#pragma unroll
        for (int v = 0; v < 8; ++v)
            C[(size_t)(rb + v) * ldc + col] = acc[j][v] + bv;
    }
}

// ---------------------------------------------------------------------------
// GRU elementwise: r,z,n gates + blend.  gi/gh are [32 x 3H].
// ---------------------------------------------------------------------------
__global__ void gru_elem_kernel(const float* __restrict__ gi, const float* __restrict__ gh,
                                const float* __restrict__ h, float* __restrict__ h2) {
    int i = blockIdx.x * blockDim.x + threadIdx.x;
    if (i >= NROW * HDIM) return;
    int r = i / HDIM, c = i % HDIM;
    const float* giR = gi + (size_t)r * 3 * HDIM;
    const float* ghR = gh + (size_t)r * 3 * HDIM;
    float rg = 1.f / (1.f + __expf(-(giR[c] + ghR[c])));
    float zg = 1.f / (1.f + __expf(-(giR[HDIM + c] + ghR[HDIM + c])));
    float ng = tanhf(giR[2 * HDIM + c] + rg * ghR[2 * HDIM + c]);
    h2[i] = (1.f - zg) * ng + zg * h[i];
}

// ---------------------------------------------------------------------------
// Per-row max and log-sum-exp over V for log_softmax.
// ---------------------------------------------------------------------------
__global__ void row_stats_kernel(const float* __restrict__ logits,
                                 float* __restrict__ mx, float* __restrict__ lse) {
    __shared__ float red[256];
    int r = blockIdx.x, tid = threadIdx.x;
    const float* row = logits + (size_t)r * VOCAB;
    float m = -3.4e38f;
    for (int v = tid; v < VOCAB; v += 256) m = fmaxf(m, row[v]);
    red[tid] = m; __syncthreads();
    for (int s = 128; s > 0; s >>= 1) {
        if (tid < s) red[tid] = fmaxf(red[tid], red[tid + s]);
        __syncthreads();
    }
    m = red[0]; __syncthreads();
    float sum = 0.f;
    for (int v = tid; v < VOCAB; v += 256) sum += __expf(row[v] - m);
    red[tid] = sum; __syncthreads();
    for (int s = 128; s > 0; s >>= 1) {
        if (tid < s) red[tid] += red[tid + s];
        __syncthreads();
    }
    if (tid == 0) { mx[r] = m; lse[r] = __logf(red[0]); }
}

// ---------------------------------------------------------------------------
// Top-8 helpers (JAX tie-break: lowest index wins on equal value).
// ---------------------------------------------------------------------------
__device__ __forceinline__ bool cand_better(float av, int ac, float bv, int bc) {
    return (av > bv) || (av == bv && ac < bc);
}
__device__ __forceinline__ void top8_insert(float v, int i, float* bv, int* bi) {
    if (!cand_better(v, i, bv[7], bi[7])) return;
    int p = 7;
    while (p > 0 && cand_better(v, i, bv[p - 1], bi[p - 1])) {
        bv[p] = bv[p - 1]; bi[p] = bi[p - 1]; --p;
    }
    bv[p] = v; bi[p] = i;
}

// Shared merge of 256 local top-8 lists -> global top-8, 8 argmax rounds.
__device__ void top8_merge(float* sv, int* si, float* rv, int* rc,
                           float* selv, int* seli) {
    int tid = threadIdx.x;
    for (int round = 0; round < 8; ++round) {
        float bv = -3.4e38f; int bc = 0x7fffffff;
        for (int j = 0; j < 8; ++j) {
            int s = tid * 8 + j;
            if (cand_better(sv[s], si[s], bv, bc)) { bv = sv[s]; bc = si[s]; }
        }
        rv[tid] = bv; rc[tid] = bc; __syncthreads();
        for (int s = 128; s > 0; s >>= 1) {
            if (tid < s && cand_better(rv[tid + s], rc[tid + s], rv[tid], rc[tid])) {
                rv[tid] = rv[tid + s]; rc[tid] = rc[tid + s];
            }
            __syncthreads();
        }
        if (tid == 0) { selv[round] = rv[0]; seli[round] = rc[0]; }
        __syncthreads();
        int wc = seli[round];
        for (int j = 0; j < 8; ++j) {
            int s = tid * 8 + j;
            if (si[s] == wc) sv[s] = -3.4e38f;   // unique candidate ids
        }
        __syncthreads();
    }
}

// ---------------------------------------------------------------------------
// Init: hidden = encoded tiled K times, last = BOS, histories zeroed.
// ---------------------------------------------------------------------------
__global__ void init_kernel(const float* __restrict__ encoded, float* __restrict__ h_cur,
                            int* __restrict__ last, float* __restrict__ lp_hist,
                            int* __restrict__ tok_hist, const int* __restrict__ bos) {
    int i = blockIdx.x * blockDim.x + threadIdx.x;
    if (i < NROW * HDIM) {
        int b = i / (NK * HDIM);
        int c = i % HDIM;
        h_cur[i] = encoded[(size_t)b * HDIM + c];
    }
    if (i < NROW * TMAX) { lp_hist[i] = 0.f; tok_hist[i] = 0; }
    if (i < NROW) last[i] = bos[0];
}

// ---------------------------------------------------------------------------
// Step 0 selection: per-batch top-8 over one row's log_softmax.
// ---------------------------------------------------------------------------
__global__ void select0_kernel(const float* __restrict__ logits, const float* __restrict__ mx,
                               const float* __restrict__ lse, const float* __restrict__ h2,
                               float* __restrict__ h_next, float* __restrict__ lp_hist,
                               int* __restrict__ tok_hist, float* __restrict__ prev_vals,
                               int* __restrict__ prev_tok, int* __restrict__ last) {
    __shared__ float sv[2048]; __shared__ int si[2048];
    __shared__ float rv[256];  __shared__ int rc[256];
    __shared__ float selv[8];  __shared__ int seli[8];
    int b = blockIdx.x, tid = threadIdx.x;
    int row = b * NK;
    float corr = mx[row] + lse[row];
    const float* lr = logits + (size_t)row * VOCAB;

    float bv[8]; int bi[8];
    for (int j = 0; j < 8; ++j) { bv[j] = -3.4e38f; bi[j] = 0x7fffffff; }
    for (int v = tid; v < VOCAB; v += 256)
        top8_insert(lr[v] - corr, v, bv, bi);
    for (int j = 0; j < 8; ++j) { sv[tid * 8 + j] = bv[j]; si[tid * 8 + j] = bi[j]; }
    __syncthreads();
    top8_merge(sv, si, rv, rc, selv, seli);

    // rows within a batch are identical at step 0: plain copy of hidden
    for (int i = tid; i < NK * HDIM; i += 256) {
        int g = b * NK * HDIM + i;
        h_next[g] = h2[g];
    }
    if (tid < NK) {
        int k = tid;
        lp_hist[(size_t)(b * NK + k) * TMAX]  = selv[k];
        tok_hist[(size_t)(b * NK + k) * TMAX] = seli[k];
        prev_vals[b * NK + k] = selv[k];
        prev_tok[b * NK + k]  = seli[k];
        last[b * NK + k]      = seli[k];
    }
}

// ---------------------------------------------------------------------------
// Steps 1..63: masked top-8 over K*V, beam reorder (hidden + histories).
// ---------------------------------------------------------------------------
__global__ void advance_kernel(const float* __restrict__ logits, const float* __restrict__ mx,
                               const float* __restrict__ lse, const float* __restrict__ h2,
                               float* __restrict__ h_next, float* __restrict__ lp_hist,
                               int* __restrict__ tok_hist, float* __restrict__ prev_vals,
                               int* __restrict__ prev_tok, int* __restrict__ last,
                               const int* __restrict__ eos_p, int step) {
    __shared__ float sv[2048]; __shared__ int si[2048];
    __shared__ float rv[256];  __shared__ int rc[256];
    __shared__ float selv[8];  __shared__ int seli[8];
    __shared__ float olp[NK * TMAX]; __shared__ int otk[NK * TMAX];
    int b = blockIdx.x, tid = threadIdx.x;
    int eos = eos_p[0];

    float bv[8]; int bi[8];
    for (int j = 0; j < 8; ++j) { bv[j] = -3.4e38f; bi[j] = 0x7fffffff; }
    for (int k = 0; k < NK; ++k) {
        int row = b * NK + k;
        float base = prev_vals[row];
        bool fin   = (prev_tok[row] == eos);
        float corr = mx[row] + lse[row];
        const float* lr = logits + (size_t)row * VOCAB;
        for (int v = tid; v < VOCAB; v += 256) {
            float val = fin ? ((v == eos) ? 0.0f : NEG_INF)
                            : base + lr[v] - corr;
            top8_insert(val, k * VOCAB + v, bv, bi);
        }
    }
    for (int j = 0; j < 8; ++j) { sv[tid * 8 + j] = bv[j]; si[tid * 8 + j] = bi[j]; }
    // snapshot histories before in-place reorder
    for (int e = tid; e < NK * TMAX; e += 256) {
        olp[e] = lp_hist[(size_t)b * NK * TMAX + e];
        otk[e] = tok_hist[(size_t)b * NK * TMAX + e];
    }
    __syncthreads();
    top8_merge(sv, si, rv, rc, selv, seli);

    // reorder hidden: h_next[b,k] = h2[b, which_k]
    for (int i = tid; i < NK * HDIM; i += 256) {
        int k = i / HDIM, c = i % HDIM;
        int which = seli[k] / VOCAB;
        h_next[(size_t)(b * NK + k) * HDIM + c] =
            h2[(size_t)(b * NK + which) * HDIM + c];
    }
    // reorder histories + append at position `step`
    for (int e = tid; e < NK * TMAX; e += 256) {
        int k = e / TMAX, t = e % TMAX;
        int which = seli[k] / VOCAB;
        int tok   = seli[k] % VOCAB;
        size_t g  = (size_t)b * NK * TMAX + e;
        lp_hist[g]  = (t == step) ? selv[k] : olp[which * TMAX + t];
        tok_hist[g] = (t == step) ? tok     : otk[which * TMAX + t];
    }
    if (tid < NK) {
        int k = tid, tok = seli[k] % VOCAB;
        prev_vals[b * NK + k] = selv[k];
        prev_tok[b * NK + k]  = tok;
        last[b * NK + k]      = tok;
    }
}

// ---------------------------------------------------------------------------
// Emit outputs: lp_hist floats then tok_hist int32 (bit-exact), flat concat.
// ---------------------------------------------------------------------------
__global__ void finalize_kernel(const float* __restrict__ lp_hist,
                                const int* __restrict__ tok_hist, float* __restrict__ out) {
    int i = blockIdx.x * blockDim.x + threadIdx.x;
    const int n = NROW * TMAX;
    if (i < n)            out[i] = lp_hist[i];
    else if (i < 2 * n)   ((int*)out)[i] = tok_hist[i - n];
}

// ---------------------------------------------------------------------------
extern "C" void kernel_launch(void* const* d_in, const int* in_sizes, int n_in,
                              void* d_out, int out_size, void* d_ws, size_t ws_size,
                              hipStream_t stream) {
    (void)in_sizes; (void)n_in; (void)out_size; (void)ws_size;
    const float* encoded   = (const float*)d_in[0];
    const float* embedding = (const float*)d_in[1];
    const float* W_ih      = (const float*)d_in[2];
    const float* W_hh      = (const float*)d_in[3];
    const float* b_ih      = (const float*)d_in[4];
    const float* b_hh      = (const float*)d_in[5];
    const float* W_out     = (const float*)d_in[6];
    const float* b_out     = (const float*)d_in[7];
    const int*   bos_p     = (const int*)d_in[8];
    const int*   eos_p     = (const int*)d_in[9];

    char* w = (char*)d_ws;
    auto grab = [&](size_t bytes) { char* p = w; w += (bytes + 255) & ~(size_t)255; return p; };
    __bf16* pWih  = (__bf16*)grab((size_t)(3 * HDIM / 16) * 16 * 32 * 16 * 2);   // nkc=16
    __bf16* pWhh  = (__bf16*)grab((size_t)(3 * HDIM / 16) * 32 * 32 * 16 * 2);   // nkc=32
    __bf16* pWout = (__bf16*)grab((size_t)(VOCAB / 16)    * 32 * 32 * 16 * 2);   // nkc=32
    __bf16* pAe   = (__bf16*)grab((size_t)2 * 16 * 32 * 16 * 2);
    __bf16* pAh   = (__bf16*)grab((size_t)2 * 32 * 32 * 16 * 2);
    __bf16* pAh2  = (__bf16*)grab((size_t)2 * 32 * 32 * 16 * 2);
    float*  gi    = (float*)grab((size_t)NROW * 3 * HDIM * 4);
    float*  gh    = (float*)grab((size_t)NROW * 3 * HDIM * 4);
    float*  hA    = (float*)grab((size_t)NROW * HDIM * 4);
    float*  hB    = (float*)grab((size_t)NROW * HDIM * 4);
    float*  h2    = (float*)grab((size_t)NROW * HDIM * 4);
    float*  logits= (float*)grab((size_t)NROW * VOCAB * 4);
    float*  mx    = (float*)grab(NROW * 4);
    float*  lse   = (float*)grab(NROW * 4);
    float*  pvals = (float*)grab(NROW * 4);
    int*    ptok  = (int*)grab(NROW * 4);
    int*    last  = (int*)grab(NROW * 4);
    float*  lp_hist = (float*)grab((size_t)NROW * TMAX * 4);
    int*    tok_hist= (int*)grab((size_t)NROW * TMAX * 4);

    // one-time (per launch) weight packing: f32 -> bf16 WMMA B-layout
    pack_b_kernel<<<384,  256, 0, stream>>>(W_ih,  pWih,  3 * HDIM, EDIM, 16);
    pack_b_kernel<<<768,  256, 0, stream>>>(W_hh,  pWhh,  3 * HDIM, HDIM, 32);
    pack_b_kernel<<<8000, 256, 0, stream>>>(W_out, pWout, VOCAB,    HDIM, 32);
    init_kernel<<<128, 256, 0, stream>>>(encoded, hA, last, lp_hist, tok_hist, bos_p);

    // wave counts: gates GEMM 2*(192/4)=96 waves -> 24 blocks of 128;
    //              vocab GEMM 2*(2000/4)=1000 waves -> 250 blocks of 128.
    float* hc = hA;
    float* hn = hB;
    for (int t = 0; t < TMAX; ++t) {
        pack_a_kernel<<<4, 256, 0, stream>>>(embedding, last, pAe, 16, EDIM);   // gather emb
        pack_a_kernel<<<8, 256, 0, stream>>>(hc, nullptr, pAh, 32, HDIM);
        gemm_wmma_kernel<<<24,  128, 0, stream>>>(pAe,  pWih,  b_ih,  gi, 3 * HDIM / 16, 16, 3 * HDIM);
        gemm_wmma_kernel<<<24,  128, 0, stream>>>(pAh,  pWhh,  b_hh,  gh, 3 * HDIM / 16, 32, 3 * HDIM);
        gru_elem_kernel<<<128, 256, 0, stream>>>(gi, gh, hc, h2);
        pack_a_kernel<<<8, 256, 0, stream>>>(h2, nullptr, pAh2, 32, HDIM);
        gemm_wmma_kernel<<<250, 128, 0, stream>>>(pAh2, pWout, b_out, logits, VOCAB / 16, 32, VOCAB);
        row_stats_kernel<<<NROW, 256, 0, stream>>>(logits, mx, lse);
        if (t == 0)
            select0_kernel<<<NB, 256, 0, stream>>>(logits, mx, lse, h2, hn,
                                                   lp_hist, tok_hist, pvals, ptok, last);
        else
            advance_kernel<<<NB, 256, 0, stream>>>(logits, mx, lse, h2, hn,
                                                   lp_hist, tok_hist, pvals, ptok, last,
                                                   eos_p, t);
        std::swap(hc, hn);
    }
    finalize_kernel<<<16, 256, 0, stream>>>(lp_hist, tok_hist, (float*)d_out);
}
